// P2B_XCorr_87179246174788
// MI455X (gfx1250) — matrix operand order repaired and checked by reading, not compiled, not run on hardware
//
#include <hip/hip_runtime.h>

// ---------------------------------------------------------------------------
// Fused P2B_XCorr for gfx1250 (MI455X, wave32, WMMA).
// All heavy math (dot GEMM, W2/W3 64x64 GEMMs, fea-layer GEMMs) runs on
// v_wmma_f32_16x16x32_f16; BN folded into f16 weights + f32 bias; the
// (B,H,M,N) intermediate never touches memory. Elementwise h1 stage uses
// packed f16 VALU (v_pk_fma_f16/v_pk_max); cos uses v_rcp_f32.
// ---------------------------------------------------------------------------

typedef _Float16 half_t;
typedef __attribute__((ext_vector_type(16))) _Float16 v16h;
typedef __attribute__((ext_vector_type(8)))  _Float16 v8h;
typedef __attribute__((ext_vector_type(2)))  _Float16 v2h;
typedef __attribute__((ext_vector_type(8)))  float    v8f;

#define B_ 8
#define F_ 128
#define M_ 256
#define N_ 1024
#define H_ 64
#define O_ 128

__device__ __constant__ float kBnEps  = 1e-5f;
__device__ __constant__ float kCosEps = 1e-8f;

// ---------------- WMMA fragment helpers (contiguous K order, A and B alike) ----------------

// A matrix 16x32 f16: lane L -> row m = L&15, hi = L>>4 selects K half.
// VGPR pair p holds K = hi*16 + 2p and +1.
__device__ __forceinline__ v16h load_A_frag(const half_t* base, int stride, int lane) {
  int m = lane & 15, hi = lane >> 4;
  v16h a;
#pragma unroll
  for (int p = 0; p < 8; ++p) {
    int kb = hi * 16 + 2 * p;
    const half_t* q = base + m * stride + kb;
    a[2 * p]     = q[0];
    a[2 * p + 1] = q[1];
  }
  return a;
}

// B matrix 32x16 f16: lane L -> col n = L&15; lanes 0-15 hold K=0..15,
// lanes 16-31 hold K=16..31 (half j -> K = hi*16 + j).
__device__ __forceinline__ v16h load_B_frag(const half_t* base, int stride, int lane) {
  int n = lane & 15, hi = lane >> 4;
  v16h b;
#pragma unroll
  for (int j = 0; j < 16; ++j) b[j] = base[n * stride + hi * 16 + j];
  return b;
}

__device__ __forceinline__ v8f wmma16(v16h a, v16h b, v8f c) {
  return __builtin_amdgcn_wmma_f32_16x16x32_f16(
      /*neg_a=*/false, a, /*neg_b=*/false, b,
      /*c_mod=*/(short)0, c, /*reuse_a=*/false, /*reuse_b=*/false);
}

__device__ __forceinline__ v8f zero8() {
  v8f z;
#pragma unroll
  for (int i = 0; i < 8; ++i) z[i] = 0.f;
  return z;
}

__device__ __forceinline__ v16h cat16(v8h lo, v8h hi) {
  return __builtin_shufflevector(lo, hi, 0, 1, 2, 3, 4, 5, 6, 7,
                                 8, 9, 10, 11, 12, 13, 14, 15);
}

// v_cvt_pk_rtz_f16_f32 with result bit-cast to _Float16x2.
__device__ __forceinline__ v2h pack2(float a, float b) {
  return __builtin_bit_cast(v2h, __builtin_amdgcn_cvt_pkrtz(a, b));
}

// ---------------------------- prep kernels ---------------------------------

__global__ void prep_const_kernel(
    const float* W1, const float* W2, const float* W3,
    const float* Wc1, const float* Wc2,
    const float* g1, const float* b1, const float* m1, const float* v1,
    const float* g2, const float* b2, const float* m2, const float* v2,
    const float* g3, const float* b3, const float* m3, const float* v3,
    const float* gc1, const float* bc1, const float* mc1, const float* vc1,
    half_t* w2f, half_t* w3f, half_t* wc1f, half_t* wc2f, half_t* w1cf,
    float* sc1, float* sh1, float* sh2, float* sh3, float* shc1) {
  __shared__ float s2s[H_], s3s[H_], scs[H_];
  int tid = threadIdx.x;
  if (tid < H_) {
    float sA = g1[tid] * rsqrtf(v1[tid] + kBnEps);
    sc1[tid]  = sA;
    sh1[tid]  = b1[tid] - m1[tid] * sA;
    w1cf[tid] = (half_t)(sA * W1[tid * (F_ + 1)]);  // column 0 of W1, BN-scaled
    float sB = g2[tid] * rsqrtf(v2[tid] + kBnEps);
    s2s[tid] = sB;  sh2[tid] = b2[tid] - m2[tid] * sB;
    float sC = g3[tid] * rsqrtf(v3[tid] + kBnEps);
    s3s[tid] = sC;  sh3[tid] = b3[tid] - m3[tid] * sC;
    float sD = gc1[tid] * rsqrtf(vc1[tid] + kBnEps);
    scs[tid] = sD;  shc1[tid] = bc1[tid] - mc1[tid] * sD;
  }
  __syncthreads();
  for (int i = tid; i < H_ * H_; i += blockDim.x) {
    int g = i / H_;
    w2f[i]  = (half_t)(s2s[g] * W2[i]);
    w3f[i]  = (half_t)(s3s[g] * W3[i]);
    wc1f[i] = (half_t)(scs[g] * Wc1[i]);
  }
  for (int i = tid; i < O_ * H_; i += blockDim.x) wc2f[i] = (half_t)Wc2[i];
}

__global__ void prep_norm_kernel(const float* tg, const float* sg,
                                 float* ntw, float* nsw) {
  int i = blockIdx.x * blockDim.x + threadIdx.x;
  if (i < B_ * M_) {
    int b = i / M_, m = i % M_;
    const float* p = tg + (size_t)(b * F_) * M_ + m;
    float acc = 0.f;
    for (int f = 0; f < F_; ++f) { float x = p[f * M_]; acc = fmaf(x, x, acc); }
    ntw[i] = sqrtf(acc);
  } else if (i < B_ * M_ + B_ * N_) {
    int j = i - B_ * M_;
    int b = j / N_, n = j % N_;
    const float* p = sg + (size_t)(b * F_) * N_ + n;
    float acc = 0.f;
    for (int f = 0; f < F_; ++f) { float x = p[f * N_]; acc = fmaf(x, x, acc); }
    nsw[j] = sqrtf(acc);
  }
}

// aa[b][m][h] = sc1[h]*(W1[:,1:]·t[b,:,m]) + sh1[h]   (f16, h contiguous)
__global__ void prep_aa_kernel(const float* tg, const float* W1,
                               const float* sc1, const float* sh1, half_t* aa) {
  int i = blockIdx.x * blockDim.x + threadIdx.x;
  int h = i % H_;
  int m = (i / H_) % M_;
  int b = i / (H_ * M_);
  const float* tp = tg + (size_t)(b * F_) * M_ + m;
  const float* wp = W1 + h * (F_ + 1) + 1;
  float acc = 0.f;
  for (int f = 0; f < F_; ++f) acc = fmaf(wp[f], tp[f * M_], acc);
  aa[i] = (half_t)fmaf(sc1[h], acc, sh1[h]);
}

// ---------------------------- fused main kernel -----------------------------
// Grid: (N/64, B). 128 threads = 4 waves; wave w owns n-columns
// [blockIdx.x*64 + w*16, +16). Loops m-tiles of 16; runs the whole
// cos -> h1 -> h2 -> h3 -> max chain in registers/WMMA; then fea-layer.

__global__ __launch_bounds__(128, 1)
void fused_xcorr_kernel(const float* __restrict__ tg, const float* __restrict__ sg,
                        const half_t* __restrict__ w2f, const half_t* __restrict__ w3f,
                        const half_t* __restrict__ wc1f, const half_t* __restrict__ wc2f,
                        const half_t* __restrict__ aa, const half_t* __restrict__ w1cf,
                        const float* __restrict__ sh2, const float* __restrict__ sh3,
                        const float* __restrict__ shc1,
                        const float* __restrict__ ntg, const float* __restrict__ nsg,
                        const float* __restrict__ bc2, float* __restrict__ out) {
  __shared__ half_t W2lds[H_ * H_];
  __shared__ half_t W3lds[H_ * H_];
  __shared__ half_t Wc1lds[H_ * H_];
  __shared__ half_t Wc2lds[O_ * H_];
  __shared__ half_t sB[4][16 * F_];      // per-wave search tile  [n][f]
  __shared__ half_t tA[4][16 * F_];      // per-wave template tile [m][f]
  __shared__ float  cosT[4][16][16];     // per-wave cos tile [m][n]
  __shared__ half_t hstage[4][16 * H_];  // per-wave staging  [n][h]

  const int tid  = threadIdx.x;
  const int wv   = tid >> 5;
  const int lane = tid & 31;
  const int n16  = lane & 15;
  const int hi   = lane >> 4;
  const int b    = blockIdx.y;
  const int n0   = blockIdx.x * 64 + wv * 16;

  // Cooperative weight load into LDS.
  for (int i = tid; i < H_ * H_; i += 128) {
    W2lds[i]  = w2f[i];
    W3lds[i]  = w3f[i];
    Wc1lds[i] = wc1f[i];
  }
  for (int i = tid; i < O_ * H_; i += 128) Wc2lds[i] = wc2f[i];

  // Per-wave search tile (f32 -> f16), [n][f] in LDS; global reads coalesced
  // over n (16 consecutive lanes hit 16 consecutive columns).
  for (int i = lane; i < 16 * F_; i += 32) {
    int n = i & 15, f = i >> 4;
    sB[wv][n * F_ + f] = (half_t)sg[((size_t)b * F_ + f) * N_ + (n0 + n)];
  }
  __syncthreads();

  // Register-resident weight fragments for the hot loop.
  v16h a2[4][2], a3[4][2];
#pragma unroll
  for (int G = 0; G < 4; ++G)
#pragma unroll
    for (int c = 0; c < 2; ++c) {
      a2[G][c] = load_A_frag(&W2lds[(16 * G) * H_ + 32 * c], H_, lane);
      a3[G][c] = load_A_frag(&W3lds[(16 * G) * H_ + 32 * c], H_, lane);
    }
  v16h bs[4];
#pragma unroll
  for (int c = 0; c < 4; ++c) bs[c] = load_B_frag(&sB[wv][32 * c], F_, lane);

  // Layer-1 constants as packed f16 (B-fragment index: h = c*32 + hi*16 + j).
  v16h w1ch[2];
#pragma unroll
  for (int c = 0; c < 2; ++c) {
    const v8h* wp = (const v8h*)__builtin_assume_aligned(w1cf + c * 32 + hi * 16, 16);
    w1ch[c] = cat16(wp[0], wp[1]);
  }
  v16h zh;
#pragma unroll
  for (int j = 0; j < 16; ++j) zh[j] = (half_t)0.f;

  // Biases in accumulator layout: g = 16G + hi*8 + r.
  float sh2r[4][8], sh3r[4][8];
#pragma unroll
  for (int G = 0; G < 4; ++G)
#pragma unroll
    for (int r = 0; r < 8; ++r) {
      sh2r[G][r] = sh2[16 * G + hi * 8 + r];
      sh3r[G][r] = sh3[16 * G + hi * 8 + r];
    }
  const float nsv = nsg[b * N_ + n0 + n16];

  v8f pmax[4];
#pragma unroll
  for (int G = 0; G < 4; ++G) pmax[G] = zero8();  // relu output >= 0

  for (int m0 = 0; m0 < M_; m0 += 16) {
    // Template tile (f32 -> f16), [m][f] in LDS; coalesced over m.
    for (int i = lane; i < 16 * F_; i += 32) {
      int mm = i & 15, f = i >> 4;
      tA[wv][mm * F_ + f] = (half_t)tg[((size_t)b * F_ + f) * M_ + (m0 + mm)];
    }
    // Prefetch next template tile (global_prefetch on gfx1250).
    if (m0 + 16 < M_) {
#pragma unroll
      for (int q = 0; q < 4; ++q)
        __builtin_prefetch(&tg[((size_t)b * F_ + (q * 32 + lane)) * M_ + m0 + 16], 0, 1);
    }
    // dot tile = t_tile^T (16x128) x s_tile (128x16), K in 4 chunks of 32.
    v8f dotacc = zero8();
#pragma unroll
    for (int c = 0; c < 4; ++c) {
      v16h at = load_A_frag(&tA[wv][32 * c], F_, lane);
      dotacc = wmma16(at, bs[c], dotacc);
    }
    // cos tile -> LDS (per-m broadcast across the h dimension comes next).
#pragma unroll
    for (int r = 0; r < 8; ++r) {
      int mm = hi * 8 + r;
      float ntv = ntg[b * M_ + m0 + mm];
      float den = fmaxf(ntv * nsv, kCosEps);
      cosT[wv][mm][n16] = dotacc[r] * __builtin_amdgcn_rcpf(den);
    }

#pragma unroll 1
    for (int mr = 0; mr < 16; ++mr) {
      const int m = m0 + mr;
      const half_t cv = (half_t)cosT[wv][mr][n16];
      const half_t* aap = aa + (size_t)(b * M_ + m) * H_;

      // h1 in B-fragment registers via packed f16: relu(w1c[h]*cos + aa[h]).
      v16h h1[2];
#pragma unroll
      for (int c = 0; c < 2; ++c) {
        const v8h* ap = (const v8h*)__builtin_assume_aligned(aap + c * 32 + hi * 16, 16);
        v16h aav = cat16(ap[0], ap[1]);
        v16h x = w1ch[c] * cv + aav;              // v_pk_fma_f16
        h1[c] = __builtin_elementwise_max(x, zh); // v_pk_max_num_f16
      }
      // h2 = relu(W2bn * h1 + sh2)
      v8f hacc[4];
#pragma unroll
      for (int G = 0; G < 4; ++G) {
        hacc[G] = zero8();
#pragma unroll
        for (int c = 0; c < 2; ++c) hacc[G] = wmma16(a2[G][c], h1[c], hacc[G]);
      }
      // D layout (g = 16G + hi*8 + r) -> LDS [n][h]; pack row pairs to b32.
#pragma unroll
      for (int G = 0; G < 4; ++G)
#pragma unroll
        for (int r = 0; r < 8; r += 2) {
          int g = 16 * G + hi * 8 + r;
          float e0 = fmaxf(hacc[G][r]     + sh2r[G][r],     0.f);
          float e1 = fmaxf(hacc[G][r + 1] + sh2r[G][r + 1], 0.f);
          *(v2h*)&hstage[wv][n16 * H_ + g] = pack2(e0, e1);
        }
      v16h h2b[2];
#pragma unroll
      for (int c = 0; c < 2; ++c) h2b[c] = load_B_frag(&hstage[wv][32 * c], H_, lane);
      // h3 = relu(W3bn * h2 + sh3); fold the max-over-m pool immediately.
#pragma unroll
      for (int G = 0; G < 4; ++G) {
        v8f h3 = zero8();
#pragma unroll
        for (int c = 0; c < 2; ++c) h3 = wmma16(a3[G][c], h2b[c], h3);
#pragma unroll
        for (int r = 0; r < 8; ++r)
          pmax[G][r] = fmaxf(pmax[G][r], fmaxf(h3[r] + sh3r[G][r], 0.f));
      }
    }
  }

  // ---- fea_layer: c = relu(Wc1bn*p + shc1); out = Wc2*c + bc2 ----
#pragma unroll
  for (int G = 0; G < 4; ++G)
#pragma unroll
    for (int r = 0; r < 8; r += 2) {
      int g = 16 * G + hi * 8 + r;
      *(v2h*)&hstage[wv][n16 * H_ + g] = pack2(pmax[G][r], pmax[G][r + 1]);
    }
  v16h pb[2];
#pragma unroll
  for (int c = 0; c < 2; ++c) pb[c] = load_B_frag(&hstage[wv][32 * c], H_, lane);

  v8f cacc[4];
#pragma unroll
  for (int G = 0; G < 4; ++G) {
    cacc[G] = zero8();
#pragma unroll
    for (int c = 0; c < 2; ++c) {
      v16h aw = load_A_frag(&Wc1lds[(16 * G) * H_ + 32 * c], H_, lane);
      cacc[G] = wmma16(aw, pb[c], cacc[G]);
    }
  }
#pragma unroll
  for (int G = 0; G < 4; ++G)
#pragma unroll
    for (int r = 0; r < 8; r += 2) {
      int g = 16 * G + hi * 8 + r;
      float e0 = fmaxf(cacc[G][r]     + shc1[g],     0.f);
      float e1 = fmaxf(cacc[G][r + 1] + shc1[g + 1], 0.f);
      *(v2h*)&hstage[wv][n16 * H_ + g] = pack2(e0, e1);
    }
  v16h cb[2];
#pragma unroll
  for (int c = 0; c < 2; ++c) cb[c] = load_B_frag(&hstage[wv][32 * c], H_, lane);

  for (int G2 = 0; G2 < 8; ++G2) {  // O = 128 -> 8 row tiles
    v8f oacc = zero8();
#pragma unroll
    for (int c = 0; c < 2; ++c) {
      v16h aw = load_A_frag(&Wc2lds[(16 * G2) * H_ + 32 * c], H_, lane);
      oacc = wmma16(aw, cb[c], oacc);
    }
#pragma unroll
    for (int r = 0; r < 8; ++r) {
      int o = 16 * G2 + hi * 8 + r;
      out[((size_t)b * O_ + o) * N_ + n0 + n16] = oacc[r] + bc2[o];
    }
  }
}

// ------------------------------- launcher -----------------------------------

extern "C" void kernel_launch(void* const* d_in, const int* in_sizes, int n_in,
                              void* d_out, int out_size, void* d_ws, size_t ws_size,
                              hipStream_t stream) {
  const float* tg  = (const float*)d_in[0];
  const float* sg  = (const float*)d_in[1];
  const float* W1  = (const float*)d_in[2];
  const float* W2  = (const float*)d_in[3];
  const float* W3  = (const float*)d_in[4];
  const float* Wc1 = (const float*)d_in[5];
  const float* Wc2 = (const float*)d_in[6];
  const float* bc2 = (const float*)d_in[7];
  const float* g1 = (const float*)d_in[8],  *b1 = (const float*)d_in[9];
  const float* m1 = (const float*)d_in[10], *v1 = (const float*)d_in[11];
  const float* g2 = (const float*)d_in[12], *b2 = (const float*)d_in[13];
  const float* m2 = (const float*)d_in[14], *v2 = (const float*)d_in[15];
  const float* g3 = (const float*)d_in[16], *b3 = (const float*)d_in[17];
  const float* m3 = (const float*)d_in[18], *v3 = (const float*)d_in[19];
  const float* gc1 = (const float*)d_in[20], *bc1 = (const float*)d_in[21];
  const float* mc1 = (const float*)d_in[22], *vc1 = (const float*)d_in[23];
  float* out = (float*)d_out;

  // Workspace carve-up (all offsets keep 16B alignment).
  char* ws = (char*)d_ws;
  half_t* w2f  = (half_t*)ws;  ws += H_ * H_ * sizeof(half_t);
  half_t* w3f  = (half_t*)ws;  ws += H_ * H_ * sizeof(half_t);
  half_t* wc1f = (half_t*)ws;  ws += H_ * H_ * sizeof(half_t);
  half_t* wc2f = (half_t*)ws;  ws += O_ * H_ * sizeof(half_t);
  half_t* w1cf = (half_t*)ws;  ws += H_ * sizeof(half_t);   // 128B, keeps 16B align
  float* sc1  = (float*)ws;    ws += H_ * sizeof(float);
  float* sh1  = (float*)ws;    ws += H_ * sizeof(float);
  float* sh2  = (float*)ws;    ws += H_ * sizeof(float);
  float* sh3  = (float*)ws;    ws += H_ * sizeof(float);
  float* shc1 = (float*)ws;    ws += H_ * sizeof(float);
  float* ntw  = (float*)ws;    ws += B_ * M_ * sizeof(float);
  float* nsw  = (float*)ws;    ws += B_ * N_ * sizeof(float);
  half_t* aa  = (half_t*)ws;   ws += (size_t)B_ * M_ * H_ * sizeof(half_t);

  prep_const_kernel<<<1, 256, 0, stream>>>(
      W1, W2, W3, Wc1, Wc2,
      g1, b1, m1, v1, g2, b2, m2, v2, g3, b3, m3, v3, gc1, bc1, mc1, vc1,
      w2f, w3f, wc1f, wc2f, w1cf, sc1, sh1, sh2, sh3, shc1);

  int norm_total = B_ * M_ + B_ * N_;
  prep_norm_kernel<<<(norm_total + 255) / 256, 256, 0, stream>>>(tg, sg, ntw, nsw);

  prep_aa_kernel<<<(B_ * M_ * H_) / 256, 256, 0, stream>>>(tg, W1, sc1, sh1, aa);

  dim3 grid(N_ / 64, B_);
  fused_xcorr_kernel<<<grid, 128, 0, stream>>>(
      tg, sg, w2f, w3f, wc1f, wc2f, aa, w1cf, sh2, sh3, shc1,
      ntw, nsw, bc2, out);
}
